// MultiHeadCosineSelfAttention_62612033241696
// MI455X (gfx1250) — matrix operand order repaired and checked by reading, not compiled, not run on hardware
//
#include <hip/hip_runtime.h>
#include <hip/hip_bf16.h>
#include <math.h>

// ---------------------------------------------------------------------------
// MultiHeadCosineSelfAttention for MI455X (gfx1250, wave32, WMMA bf16)
//
// out = (Q K^T / c) V == Q ((K^T V)/c): attention collapses to a 64x64
// per-(b,h) matrix.  Heavy GEMMs run on v_wmma_f32_16x16x32_bf16 with
// async global->LDS B-tile staging (double buffered, ASYNCcnt-tracked).
// ---------------------------------------------------------------------------

typedef __attribute__((ext_vector_type(16))) __bf16 v16bf;
typedef __attribute__((ext_vector_type(8)))  float  v8f;

#define FEATS  1024
#define NHEAD  16
#define HDIM   64
#define SEQ    2048
#define NBATCH 4
#define ROWS   (NBATCH * SEQ)   // 8192

union F8 { uint4 u4; __bf16 h[8]; };

enum { EPI_Q = 0, EPI_K = 1, EPI_V = 2, EPI_F32 = 3, EPI_OBF = 4, EPI_FINAL = 5 };

// CDNA5 async copy: 16B per lane, global -> LDS, tracked by ASYNCcnt.
__device__ __forceinline__ void async_b128_to_lds(unsigned lds_off,
                                                  const __bf16* gptr) {
    asm volatile("global_load_async_to_lds_b128 %0, %1, off"
                 :: "v"(lds_off), "v"((unsigned long long)(size_t)gptr)
                 : "memory");
}
__device__ __forceinline__ void wait_async0() {
    asm volatile("s_wait_asynccnt 0x0" ::: "memory");
}

// Build a WMMA 16-bf16 fragment from two 16-byte chunks by pure bitcast:
// elements 0-7 = first b128, elements 8-15 = second b128 (no per-element ops,
// so this stays entirely in registers).
__device__ __forceinline__ v16bf ld_frag2(const __bf16* p0, const __bf16* p1) {
    union { uint4 u[2]; v16bf v; } t;
    t.u[0] = *(const uint4*)p0;
    t.u[1] = *(const uint4*)p1;
    return t.v;
}

// Generic bf16 WMMA GEMM:  C[M x N] = A[M x K] * Bt[N x K]^T  (Bt = B^T, row-major)
// Block: 256 threads = 8 waves; each wave owns a 16(M) x 64(N) strip -> 4 accumulators.
// Bt K-tile (64 x 32 bf16) is async-staged into double-buffered LDS.
template<int EPI>
__global__ __launch_bounds__(256) void gemm_bf16(
    const __bf16* __restrict__ A,  int lda, long long sA,
    const __bf16* __restrict__ Bt, int ldb, long long sB,
    void* __restrict__ Cv, long long sC,
    const float* __restrict__ bias,
    int M, int N, int Kdim)
{
    __shared__ __align__(16) __bf16 Bs[2][64 * 40];   // padded row stride 40

    const int tid  = threadIdx.x;
    const int lane = tid & 31;
    const int wave = tid >> 5;
    const int lh   = lane & 15;
    const int hi   = lane >> 4;
    const int z    = blockIdx.z;

    A  += (size_t)z * sA;
    Bt += (size_t)z * sB;

    const int N0 = blockIdx.y * 64;
    const int M0 = blockIdx.x * 128 + wave * 16;

    // A fragment row for this lane (ISA 7.12.2: lanes 0-15 -> M=lane, 16-31 -> M=lane-16)
    const int arow = M0 + lh;
    const __bf16* Aptr = A + (size_t)(arow < M ? arow : 0) * lda;

    // staging assignment: thread -> (Bt row 0..63, 8-element column group)
    const int sr = tid >> 2;
    const int sc = (tid & 3) * 8;
    const __bf16* Bsrc = Bt + (size_t)(N0 + sr) * ldb + sc;

    unsigned lds_dst[2];
    lds_dst[0] = (unsigned)(size_t)&Bs[0][sr * 40 + sc];
    lds_dst[1] = (unsigned)(size_t)&Bs[1][sr * 40 + sc];

    v8f acc[4];
#pragma unroll
    for (int j = 0; j < 4; ++j)
#pragma unroll
        for (int e = 0; e < 8; ++e) acc[j][e] = 0.f;

    const int nK = Kdim >> 5;
    // prologue: stage first B tile
    async_b128_to_lds(lds_dst[0], Bsrc);

    int buf = 0;
    for (int kb = 0; kb < nK; ++kb) {
        const int K0 = kb << 5;
        wait_async0();          // my slice of tile kb landed in LDS
        __syncthreads();        // everyone's slice landed; prev-buf readers done

        if (kb + 1 < nK)        // stage tile kb+1 into the other buffer
            async_b128_to_lds(lds_dst[buf ^ 1], Bsrc + K0 + 32);

        // A fragment: elements 0-15 = K{hi*16..+15}
        v16bf af = ld_frag2(Aptr + K0 + hi * 16, Aptr + K0 + hi * 16 + 8);

        if (kb + 1 < nK)
            __builtin_prefetch((const void*)(Aptr + K0 + 32), 0, 3);

        // all four B fragments first (batched ds_load_b128), then WMMA burst
        const __bf16* bbase = &Bs[buf][(size_t)lh * 40 + hi * 16];
        v16bf bf0 = ld_frag2(bbase,           bbase + 8);
        v16bf bf1 = ld_frag2(bbase + 16 * 40, bbase + 16 * 40 + 8);
        v16bf bf2 = ld_frag2(bbase + 32 * 40, bbase + 32 * 40 + 8);
        v16bf bf3 = ld_frag2(bbase + 48 * 40, bbase + 48 * 40 + 8);

        acc[0] = __builtin_amdgcn_wmma_f32_16x16x32_bf16(false, af, false, bf0, (short)0, acc[0], false, false);
        acc[1] = __builtin_amdgcn_wmma_f32_16x16x32_bf16(false, af, false, bf1, (short)0, acc[1], false, false);
        acc[2] = __builtin_amdgcn_wmma_f32_16x16x32_bf16(false, af, false, bf2, (short)0, acc[2], false, false);
        acc[3] = __builtin_amdgcn_wmma_f32_16x16x32_bf16(false, af, false, bf3, (short)0, acc[3], false, false);

        buf ^= 1;
    }

    // ---- epilogue -------------------------------------------------------
    // C layout: element e of acc[j] -> row M0 + hi*8 + e, col N0 + j*16 + lh
    const int row0 = M0 + hi * 8;

    if (EPI == EPI_Q || EPI == EPI_K || EPI == EPI_V) {
#pragma unroll
        for (int j = 0; j < 4; ++j) {
            float bb = bias[N0 + j * 16 + lh];
#pragma unroll
            for (int e = 0; e < 8; ++e) acc[j][e] += bb;
        }
    }

    float inv[8];
#pragma unroll
    for (int e = 0; e < 8; ++e) inv[e] = 1.f;
    if (EPI == EPI_Q || EPI == EPI_K) {
        // per-row L2 norm over this head's 64 columns: 4 in-lane partials,
        // then butterfly across the 16 lanes of the same half-wave.
#pragma unroll
        for (int e = 0; e < 8; ++e) {
            float ss = 0.f;
#pragma unroll
            for (int j = 0; j < 4; ++j) ss += acc[j][e] * acc[j][e];
            for (int msk = 1; msk < 16; msk <<= 1) ss += __shfl_xor(ss, msk, 32);
            inv[e] = 1.f / fmaxf(sqrtf(ss), 1e-12f);
        }
    }

    const int b  = row0 >> 11;        // row / 2048
    const int n0 = row0 & 2047;
    const int h  = N0 >> 6;

    if (EPI == EPI_Q) {
        // Q[b,h][n][d] row-major per head
        __bf16* dst = (__bf16*)Cv + (size_t)(b * NHEAD + h) * SEQ * HDIM;
#pragma unroll
        for (int j = 0; j < 4; ++j) {
            int d = j * 16 + lh;
#pragma unroll
            for (int e = 0; e < 8; ++e)
                dst[(size_t)(n0 + e) * HDIM + d] = (__bf16)(acc[j][e] * inv[e]);
        }
    } else if (EPI == EPI_K || EPI == EPI_V) {
        // transposed Kt/Vt[b,h][d][n]: 8 accumulator rows are contiguous n
        // -> single 16-byte store per accumulator.
        __bf16* dst = (__bf16*)Cv + (size_t)(b * NHEAD + h) * HDIM * SEQ;
#pragma unroll
        for (int j = 0; j < 4; ++j) {
            int d = j * 16 + lh;
            F8 p;
#pragma unroll
            for (int e = 0; e < 8; ++e)
                p.h[e] = (__bf16)(EPI == EPI_K ? acc[j][e] * inv[e] : acc[j][e]);
            *(uint4*)(dst + (size_t)d * SEQ + n0) = p.u4;
        }
    } else if (EPI == EPI_F32) {
        float* dst = (float*)Cv + (size_t)z * sC;
#pragma unroll
        for (int j = 0; j < 4; ++j) {
            int c = N0 + j * 16 + lh;
            if (c < N)
#pragma unroll
                for (int e = 0; e < 8; ++e) {
                    int r = row0 + e;
                    if (r < M) dst[(size_t)r * N + c] = acc[j][e];
                }
        }
    } else if (EPI == EPI_OBF) {
        // write O back into [b][n][h*64+d] (bf16) for the final projection
        int bb2 = z >> 4, hh = z & 15;
        __bf16* dst = (__bf16*)Cv + (size_t)bb2 * SEQ * FEATS + hh * HDIM;
#pragma unroll
        for (int j = 0; j < 4; ++j) {
            int d = j * 16 + lh;
#pragma unroll
            for (int e = 0; e < 8; ++e) {
                int r = row0 + e;
                if (r < M) dst[(size_t)r * FEATS + d] = (__bf16)acc[j][e];
            }
        }
    } else if (EPI == EPI_FINAL) {
        float* dst = (float*)Cv;
#pragma unroll
        for (int j = 0; j < 4; ++j) {
            int c = N0 + j * 16 + lh;
            float bb = bias[c];
#pragma unroll
            for (int e = 0; e < 8; ++e) {
                int r = row0 + e;
                if (r < M) dst[(size_t)r * N + c] = acc[j][e] + bb;
            }
        }
    }
}

// fp32 -> bf16 straight copy
__global__ void k_f32_to_bf16(const float* __restrict__ in, __bf16* __restrict__ out, int n) {
    int i = blockIdx.x * blockDim.x + threadIdx.x;
    if (i < n) out[i] = (__bf16)in[i];
}

// W[K][N] fp32 -> Wt[N][K] bf16 (LDS tiled transpose)
__global__ __launch_bounds__(256) void k_transpose_w(const float* __restrict__ in,
                                                     __bf16* __restrict__ out) {
    __shared__ float tile[32][33];
    int tx = threadIdx.x, ty = threadIdx.y;
    int x  = blockIdx.x * 32 + tx;
    int y0 = blockIdx.y * 32;
    for (int i = ty; i < 32; i += 8)
        tile[i][tx] = in[(size_t)(y0 + i) * FEATS + x];
    __syncthreads();
    int xo = y0 + tx;
    int yo = blockIdx.x * 32;
    for (int i = ty; i < 32; i += 8)
        out[(size_t)(yo + i) * FEATS + xo] = (__bf16)tile[tx][i];
}

// T[b,h] = (V^T K); fold per-head scale 1/n^{sigmoid(m_h)} = 2^{-11*sigmoid(m_h)}
__global__ void k_scale_T(const float* __restrict__ T, const float* __restrict__ m,
                          __bf16* __restrict__ Tb) {
    int i = blockIdx.x * 256 + threadIdx.x;     // 64 * 64 * 64 total
    int z = i >> 12;
    int h = z & 15;
    float s   = 1.f / (1.f + __expf(-m[h]));
    float inv = exp2f(-11.f * s);               // n = 2048 = 2^11
    Tb[i] = (__bf16)(T[i] * inv);
}

extern "C" void kernel_launch(void* const* d_in, const int* in_sizes, int n_in,
                              void* d_out, int out_size, void* d_ws, size_t ws_size,
                              hipStream_t stream) {
    (void)in_sizes; (void)n_in; (void)out_size; (void)ws_size;

    const float* x  = (const float*)d_in[0];
    const float* Wq = (const float*)d_in[1];
    const float* bq = (const float*)d_in[2];
    const float* Wk = (const float*)d_in[3];
    const float* bk = (const float*)d_in[4];
    const float* Wv = (const float*)d_in[5];
    const float* bv = (const float*)d_in[6];
    const float* Wo = (const float*)d_in[7];
    const float* bo = (const float*)d_in[8];
    const float* m  = (const float*)d_in[9];
    float* out = (float*)d_out;

    char* ws = (char*)d_ws;
    size_t off = 0;
    auto alloc = [&](size_t bytes) -> void* {
        void* p = ws + off;
        off += (bytes + 255) & ~(size_t)255;
        return p;
    };

    __bf16* xbf = (__bf16*)alloc((size_t)ROWS * FEATS * 2);
    __bf16* Wtq = (__bf16*)alloc((size_t)FEATS * FEATS * 2);
    __bf16* Wtk = (__bf16*)alloc((size_t)FEATS * FEATS * 2);
    __bf16* Wtv = (__bf16*)alloc((size_t)FEATS * FEATS * 2);
    __bf16* Wto = (__bf16*)alloc((size_t)FEATS * FEATS * 2);
    __bf16* Qbf = (__bf16*)alloc((size_t)ROWS * FEATS * 2);   // [b,h][n][d]
    __bf16* Ktb = (__bf16*)alloc((size_t)ROWS * FEATS * 2);   // [b,h][d][n]
    __bf16* Vtb = (__bf16*)alloc((size_t)ROWS * FEATS * 2);   // [b,h][d][n]
    float*  Tf  = (float*) alloc((size_t)NBATCH * NHEAD * HDIM * HDIM * 4);
    __bf16* Tb  = (__bf16*)alloc((size_t)NBATCH * NHEAD * HDIM * HDIM * 2);
    __bf16* Obf = (__bf16*)alloc((size_t)ROWS * FEATS * 2);   // [b][n][h*64+d]

    // 1) inputs -> bf16 (weights transposed so B-columns are contiguous)
    k_f32_to_bf16<<<(ROWS * FEATS + 255) / 256, 256, 0, stream>>>(x, xbf, ROWS * FEATS);
    dim3 tb(32, 8), tg(32, 32);
    k_transpose_w<<<tg, tb, 0, stream>>>(Wq, Wtq);
    k_transpose_w<<<tg, tb, 0, stream>>>(Wk, Wtk);
    k_transpose_w<<<tg, tb, 0, stream>>>(Wv, Wtv);
    k_transpose_w<<<tg, tb, 0, stream>>>(Wo, Wto);

    // 2) fused QKV projections (+bias, +per-head L2 norm for Q/K, layout shuffles)
    dim3 blk(256);
    dim3 gproj(ROWS / 128, FEATS / 64, 1);
    gemm_bf16<EPI_Q><<<gproj, blk, 0, stream>>>(xbf, FEATS, 0, Wtq, FEATS, 0, Qbf, 0, bq, ROWS, FEATS, FEATS);
    gemm_bf16<EPI_K><<<gproj, blk, 0, stream>>>(xbf, FEATS, 0, Wtk, FEATS, 0, Ktb, 0, bk, ROWS, FEATS, FEATS);
    gemm_bf16<EPI_V><<<gproj, blk, 0, stream>>>(xbf, FEATS, 0, Wtv, FEATS, 0, Vtb, 0, bv, ROWS, FEATS, FEATS);

    // 3) T[b,h] = V^T K   (64x64, K-dim = 2048), batched over 64 (b,h)
    gemm_bf16<EPI_F32><<<dim3(1, 1, NBATCH * NHEAD), blk, 0, stream>>>(
        Vtb, SEQ, (long long)HDIM * SEQ,
        Ktb, SEQ, (long long)HDIM * SEQ,
        Tf, (long long)HDIM * HDIM, nullptr, HDIM, HDIM, SEQ);

    // 4) scale by 1/n^{sigmoid(m_h)} and convert to bf16 (Tb = S^T = B^T for next GEMM)
    k_scale_T<<<(NBATCH * NHEAD * HDIM * HDIM) / 256, 256, 0, stream>>>(Tf, m, Tb);

    // 5) O[b,h] = Q[b,h] @ S_h  -> written into [b][n][h*64+d] bf16
    gemm_bf16<EPI_OBF><<<dim3(SEQ / 128, 1, NBATCH * NHEAD), blk, 0, stream>>>(
        Qbf, HDIM, (long long)SEQ * HDIM,
        Tb, HDIM, (long long)HDIM * HDIM,
        Obf, 0, nullptr, SEQ, HDIM, HDIM);

    // 6) out = O @ Wo + bo (fp32)
    gemm_bf16<EPI_FINAL><<<gproj, blk, 0, stream>>>(
        Obf, FEATS, 0, Wto, FEATS, 0, out, 0, bo, ROWS, FEATS, FEATS);
}
